// MultiHeadAttention_11596411699978
// MI455X (gfx1250) — matrix operand order repaired and checked by reading, not compiled, not run on hardware
//
#include <hip/hip_runtime.h>
#include <hip/hip_bf16.h>
#include <stdint.h>

typedef _Float16 half8 __attribute__((ext_vector_type(8)));
typedef _Float16 v16h  __attribute__((ext_vector_type(16)));
typedef float    v8f   __attribute__((ext_vector_type(8)));

#define B_SZ     2
#define S_LEN    2048
#define D_MODEL  2048
#define N_HEAD   16
#define HD       128

// ---------------------------------------------------------------------------
// CDNA5 async global->LDS copy (ASYNCcnt-tracked), inline asm for portability
// across ROCm-7.2 / amdgpu-toolchain builtin arity differences.
// lds_off = wave-relative LDS byte offset (low 32 bits of generic LDS pointer)
// ---------------------------------------------------------------------------
__device__ __forceinline__ void async_copy_b128(uint32_t lds_off, const void* gptr) {
  asm volatile("global_load_async_to_lds_b128 %0, %1, off"
               :: "v"(lds_off), "v"((uint64_t)(uintptr_t)gptr)
               : "memory");
}
__device__ __forceinline__ void wait_async0() {
  asm volatile("s_wait_asynccnt 0x0" ::: "memory");
}

// ---------------------------------------------------------------------------
// fp32 -> f16 conversion, 8 elements/thread (2x b128 load, 1x b128 store)
// ---------------------------------------------------------------------------
__global__ __launch_bounds__(256) void cvt_f32_f16(const float4* __restrict__ in,
                                                   half8* __restrict__ out,
                                                   int n8) {
  int i = blockIdx.x * blockDim.x + threadIdx.x;
  if (i < n8) {
    float4 a = in[2 * i];
    float4 b = in[2 * i + 1];
    half8 h = {(_Float16)a.x, (_Float16)a.y, (_Float16)a.z, (_Float16)a.w,
               (_Float16)b.x, (_Float16)b.y, (_Float16)b.z, (_Float16)b.w};
    out[i] = h;
  }
}

// ---------------------------------------------------------------------------
// Tiled WMMA GEMM: C[M,N] = A[M,K] * B[K,N], f16 in, f32 accumulate.
// Block tile 256x128, BK=32. 256 threads = 8 waves arranged 4(m) x 2(n);
// each wave computes 64x64 = 4x4 WMMA tiles (16 wmma / K-step).
// Double-buffered LDS; A tiles via async global->LDS (ASYNCcnt), B tiles
// transposed through registers.
// ---------------------------------------------------------------------------
template <bool StoreHalf>
__global__ __launch_bounds__(256)
void gemm_f16_wmma(const _Float16* __restrict__ A, const _Float16* __restrict__ B,
                   void* __restrict__ Cp, int M, int N, int K) {
  __shared__ __attribute__((aligned(16))) _Float16 Ash[2][256][48];  // [m][k]
  __shared__ __attribute__((aligned(16))) _Float16 Bts[2][128][48];  // [n][k]

  const int tid  = threadIdx.x;
  const int lane = tid & 31;
  const int wave = tid >> 5;
  const int wm   = wave >> 1;          // 0..3
  const int wn   = wave & 1;           // 0..1
  const int m0   = blockIdx.y * 256;
  const int n0   = blockIdx.x * 128;
  const int hsel = lane >> 4;          // 0: lanes 0-15, 1: lanes 16-31
  const int l15  = lane & 15;

  v8f acc[4][4];
#pragma unroll
  for (int i = 0; i < 4; ++i)
#pragma unroll
    for (int j = 0; j < 4; ++j) acc[i][j] = (v8f)0.0f;

  auto load_tiles = [&](int kt, int buf) {
    const int k0 = kt * 32;
    // A tile 256x32: 1024 b128 chunks, 4 per thread, async -> LDS
#pragma unroll
    for (int c = tid; c < 1024; c += 256) {
      int mm = c >> 2;
      int kk = (c & 3) << 3;
      async_copy_b128((uint32_t)(uintptr_t)&Ash[buf][mm][kk],
                      &A[(size_t)(m0 + mm) * K + k0 + kk]);
    }
    // B tile 32x128 -> transposed [n][k] through registers
#pragma unroll
    for (int c = tid; c < 512; c += 256) {
      int kk = c >> 4;
      int nn = (c & 15) << 3;
      half8 bv = *(const half8*)&B[(size_t)(k0 + kk) * N + n0 + nn];
#pragma unroll
      for (int j = 0; j < 8; ++j) Bts[buf][nn + j][kk] = bv[j];
    }
  };

  load_tiles(0, 0);
  wait_async0();
  __syncthreads();

  const int KT = K >> 5;
  for (int kt = 0; kt < KT; ++kt) {
    const int cur = kt & 1;
    if (kt + 1 < KT) load_tiles(kt + 1, cur ^ 1);

    // --- fragments from current buffer ---
    v16h afrag[4];
#pragma unroll
    for (int i = 0; i < 4; ++i) {
      int mrow = wm * 64 + i * 16 + l15;
      int kb   = hsel * 8;   // A layout: runs [kb,kb+8) and [kb+16,kb+24)
      *(half8*)&afrag[i]       = *(const half8*)&Ash[cur][mrow][kb];
      *((half8*)&afrag[i] + 1) = *(const half8*)&Ash[cur][mrow][kb + 16];
    }
    v16h bfrag[4];
#pragma unroll
    for (int j = 0; j < 4; ++j) {
      int nrow = wn * 64 + j * 16 + l15;
      int kb   = hsel * 8;   // B layout: runs [kb,kb+8) and [kb+16,kb+24)
      *(half8*)&bfrag[j]       = *(const half8*)&Bts[cur][nrow][kb];
      *((half8*)&bfrag[j] + 1) = *(const half8*)&Bts[cur][nrow][kb + 16];
    }
#pragma unroll
    for (int i = 0; i < 4; ++i)
#pragma unroll
      for (int j = 0; j < 4; ++j)
        acc[i][j] = __builtin_amdgcn_wmma_f32_16x16x32_f16(
            false, afrag[i], false, bfrag[j], (short)0, acc[i][j], false, false);

    wait_async0();
    __syncthreads();
  }

  // --- store (C layout: VGPR r -> row r (+8 upper half), col = lane&15) ---
#pragma unroll
  for (int i = 0; i < 4; ++i)
#pragma unroll
    for (int j = 0; j < 4; ++j) {
      int col = n0 + wn * 64 + j * 16 + l15;
#pragma unroll
      for (int r = 0; r < 8; ++r) {
        int row = m0 + wm * 64 + i * 16 + r + hsel * 8;
        if (StoreHalf)
          ((_Float16*)Cp)[(size_t)row * N + col] = (_Float16)acc[i][j][r];
        else
          ((float*)Cp)[(size_t)row * N + col] = acc[i][j][r];
      }
    }
}

// ---------------------------------------------------------------------------
// RoPE + split qkv[b,s,3,H,hd] (f16) -> Qr/Kr/Vr [b,h,s,hd] (f16)
// ---------------------------------------------------------------------------
__global__ __launch_bounds__(256)
void rope_split(const _Float16* __restrict__ qkvh, _Float16* __restrict__ Qr,
                _Float16* __restrict__ Kr, _Float16* __restrict__ Vr) {
  int idx = blockIdx.x * blockDim.x + threadIdx.x;   // 2^22 threads total
  int i   = idx & 63;
  int s   = (idx >> 6) & (S_LEN - 1);
  int bh  = idx >> 17;                               // b*16+h
  int h   = bh & (N_HEAD - 1);

  const size_t inrow = ((size_t)((bh >> 4) * S_LEN + s)) * (3 * D_MODEL);
  const size_t obase = ((size_t)bh) * S_LEN * HD + (size_t)s * HD;

  float ang = (float)s * __powf(10000.0f, -(float)i / 64.0f);
  float sn, cs;
  __sincosf(ang, &sn, &cs);

  float q1 = (float)qkvh[inrow + h * HD + i];
  float q2 = (float)qkvh[inrow + h * HD + i + 64];
  Qr[obase + i]      = (_Float16)(q1 * cs - q2 * sn);
  Qr[obase + i + 64] = (_Float16)(q2 * cs + q1 * sn);

  float k1 = (float)qkvh[inrow + D_MODEL + h * HD + i];
  float k2 = (float)qkvh[inrow + D_MODEL + h * HD + i + 64];
  Kr[obase + i]      = (_Float16)(k1 * cs - k2 * sn);
  Kr[obase + i + 64] = (_Float16)(k2 * cs + k1 * sn);

  Vr[obase + i]      = qkvh[inrow + 2 * D_MODEL + h * HD + i];
  Vr[obase + i + 64] = qkvh[inrow + 2 * D_MODEL + h * HD + i + 64];
}

// ---------------------------------------------------------------------------
// Flash attention (causal): one WG = (b,h, 64 query rows), 4 waves, each wave
// owns 16 query rows. Online softmax; WMMA for S=Q*K^T and O=P*V.
// K tiles arrive via CDNA5 async global->LDS; V transposed through registers.
// ---------------------------------------------------------------------------
__global__ __launch_bounds__(128)
void flash_attn(const _Float16* __restrict__ Q, const _Float16* __restrict__ Kb,
                const _Float16* __restrict__ Vb, _Float16* __restrict__ Obuf) {
  __shared__ __attribute__((aligned(16))) _Float16 Ksh[64][144];    // [n][d]
  __shared__ __attribute__((aligned(16))) _Float16 Vts[128][80];    // [d][n]
  __shared__ __attribute__((aligned(16))) _Float16 Psh[4][16][80];  // per-wave P

  const int bh   = blockIdx.y;          // b*16 + h
  const int b    = bh >> 4;
  const int h    = bh & 15;
  const int mblk = blockIdx.x;          // 64-row query block
  const int tid  = threadIdx.x;
  const int lane = tid & 31;
  const int wave = tid >> 5;
  const int hsel = lane >> 4;
  const int l15  = lane & 15;

  const size_t base  = (size_t)bh * S_LEN * HD;
  const int    m_row = mblk * 64 + wave * 16 + l15;      // A-frag row
  const int    erow0 = mblk * 64 + wave * 16 + hsel * 8; // C-layout row base

  // Q fragments live in registers for the whole kernel (4 K-steps of 32)
  v16h qf[4];
  {
    const _Float16* qp = Q + base + (size_t)m_row * HD;
#pragma unroll
    for (int ks = 0; ks < 4; ++ks) {
      int d0 = ks * 32 + hsel * 8;
      *(half8*)&qf[ks]       = *(const half8*)&qp[d0];
      *((half8*)&qf[ks] + 1) = *(const half8*)&qp[d0 + 16];
    }
  }

  v8f oacc[8];
#pragma unroll
  for (int t = 0; t < 8; ++t) oacc[t] = (v8f)0.0f;
  float mstat[8], lstat[8];
#pragma unroll
  for (int r = 0; r < 8; ++r) { mstat[r] = -1e30f; lstat[r] = 0.0f; }

  const float scale   = 0.08838834764831845f;  // 1/sqrt(128)
  const int   nblocks = mblk + 1;              // causal key-block limit

  for (int nb = 0; nb < nblocks; ++nb) {
    const int nbase = nb * 64;
    // ---- K tile via async global->LDS (8 b128 per thread) ----
#pragma unroll
    for (int c = tid; c < 1024; c += 128) {
      int nn = c >> 4, dc = (c & 15) << 3;
      async_copy_b128((uint32_t)(uintptr_t)&Ksh[nn][dc],
                      &Kb[base + (size_t)(nbase + nn) * HD + dc]);
    }
    // ---- V tile transposed through registers ----
#pragma unroll
    for (int c = tid; c < 1024; c += 128) {
      int nn = c >> 4, dc = (c & 15) << 3;
      half8 vv = *(const half8*)&Vb[base + (size_t)(nbase + nn) * HD + dc];
#pragma unroll
      for (int j = 0; j < 8; ++j) Vts[dc + j][nn] = vv[j];
    }
    wait_async0();
    __syncthreads();

    // ---- S = Q * K^T (4 n-tiles of 16) ----
    v8f sacc[4];
#pragma unroll
    for (int nt = 0; nt < 4; ++nt) {
      sacc[nt] = (v8f)0.0f;
#pragma unroll
      for (int ks = 0; ks < 4; ++ks) {
        v16h bf;
        int nrow = nt * 16 + l15;
        int d0   = ks * 32 + hsel * 8;
        *(half8*)&bf       = *(const half8*)&Ksh[nrow][d0];
        *((half8*)&bf + 1) = *(const half8*)&Ksh[nrow][d0 + 16];
        sacc[nt] = __builtin_amdgcn_wmma_f32_16x16x32_f16(
            false, qf[ks], false, bf, (short)0, sacc[nt], false, false);
      }
    }

    // ---- scale + causal mask ----
#pragma unroll
    for (int nt = 0; nt < 4; ++nt) {
      int col = nbase + nt * 16 + l15;
#pragma unroll
      for (int r = 0; r < 8; ++r) {
        float v = sacc[nt][r] * scale;
        sacc[nt][r] = (col > erow0 + r) ? -1.0e9f : v;
      }
    }

    // ---- online softmax (row stats replicated across each 16-lane half) ----
    float mnew[8];
#pragma unroll
    for (int r = 0; r < 8; ++r) {
      float mx = mstat[r];
#pragma unroll
      for (int nt = 0; nt < 4; ++nt) mx = fmaxf(mx, sacc[nt][r]);
      for (int off = 1; off < 16; off <<= 1) mx = fmaxf(mx, __shfl_xor(mx, off, 32));
      mnew[r] = mx;
    }
#pragma unroll
    for (int r = 0; r < 8; ++r) {
      float alpha = __expf(mstat[r] - mnew[r]);
#pragma unroll
      for (int t = 0; t < 8; ++t) oacc[t][r] *= alpha;
      lstat[r] *= alpha;
      mstat[r] = mnew[r];
    }
#pragma unroll
    for (int r = 0; r < 8; ++r) {
      float rs = 0.0f;
#pragma unroll
      for (int nt = 0; nt < 4; ++nt) {
        float p = __expf(sacc[nt][r] - mnew[r]);
        sacc[nt][r] = p;
        rs += p;
      }
      for (int off = 1; off < 16; off <<= 1) rs += __shfl_xor(rs, off, 32);
      lstat[r] += rs;
    }

    // ---- P -> LDS (C-layout scatter), reload as A fragments ----
#pragma unroll
    for (int nt = 0; nt < 4; ++nt)
#pragma unroll
      for (int r = 0; r < 8; ++r)
        Psh[wave][r + hsel * 8][nt * 16 + l15] = (_Float16)sacc[nt][r];
    __builtin_amdgcn_wave_barrier();

    v16h pf[2];
#pragma unroll
    for (int ks = 0; ks < 2; ++ks) {
      int c0 = ks * 32 + hsel * 8;
      *(half8*)&pf[ks]       = *(const half8*)&Psh[wave][l15][c0];
      *((half8*)&pf[ks] + 1) = *(const half8*)&Psh[wave][l15][c0 + 16];
    }

    // ---- O += P * V ----
#pragma unroll
    for (int dt = 0; dt < 8; ++dt) {
      int drow = dt * 16 + l15;
#pragma unroll
      for (int ks = 0; ks < 2; ++ks) {
        v16h bf;
        int c0 = ks * 32 + hsel * 8;
        *(half8*)&bf       = *(const half8*)&Vts[drow][c0];
        *((half8*)&bf + 1) = *(const half8*)&Vts[drow][c0 + 16];
        oacc[dt] = __builtin_amdgcn_wmma_f32_16x16x32_f16(
            false, pf[ks], false, bf, (short)0, oacc[dt], false, false);
      }
    }
    __syncthreads();
  }

  // ---- normalize + store to Obuf[b, s, h*HD + d] ----
#pragma unroll
  for (int r = 0; r < 8; ++r) {
    float inv = 1.0f / lstat[r];
    int row = erow0 + r;
#pragma unroll
    for (int dt = 0; dt < 8; ++dt) {
      int col = h * HD + dt * 16 + l15;
      Obuf[((size_t)(b * S_LEN + row)) * D_MODEL + col] =
          (_Float16)(oacc[dt][r] * inv);
    }
  }
}

// ---------------------------------------------------------------------------
// host-side orchestration
// ---------------------------------------------------------------------------
extern "C" void kernel_launch(void* const* d_in, const int* in_sizes, int n_in,
                              void* d_out, int out_size, void* d_ws, size_t ws_size,
                              hipStream_t stream) {
  const float* x  = (const float*)d_in[0];
  const float* Wp = (const float*)d_in[1];
  const float* Wo = (const float*)d_in[2];
  float* out = (float*)d_out;

  const size_t BS = (size_t)B_SZ * S_LEN;  // 4096 rows
  char* ws = (char*)d_ws;
  size_t off = 0;
  _Float16* xh   = (_Float16*)(ws + off); off += BS * D_MODEL * 2;
  _Float16* Wph  = (_Float16*)(ws + off); off += (size_t)D_MODEL * 3 * D_MODEL * 2;
  _Float16* Woh  = (_Float16*)(ws + off); off += (size_t)D_MODEL * D_MODEL * 2;
  _Float16* qkvh = (_Float16*)(ws + off); off += BS * 3 * D_MODEL * 2;
  _Float16* Qr   = (_Float16*)(ws + off); off += BS * D_MODEL * 2;
  _Float16* Kr   = (_Float16*)(ws + off); off += BS * D_MODEL * 2;
  _Float16* Vr   = (_Float16*)(ws + off); off += BS * D_MODEL * 2;
  _Float16* Obuf = (_Float16*)(ws + off); off += BS * D_MODEL * 2;

  // 1) fp32 -> f16 (8 elems / thread)
  {
    int n1 = (int)(BS * D_MODEL) / 8;
    int n2 = (D_MODEL * 3 * D_MODEL) / 8;
    int n3 = (D_MODEL * D_MODEL) / 8;
    cvt_f32_f16<<<(n1 + 255) / 256, 256, 0, stream>>>((const float4*)x, (half8*)xh, n1);
    cvt_f32_f16<<<(n2 + 255) / 256, 256, 0, stream>>>((const float4*)Wp, (half8*)Wph, n2);
    cvt_f32_f16<<<(n3 + 255) / 256, 256, 0, stream>>>((const float4*)Wo, (half8*)Woh, n3);
  }
  // 2) qkv = x @ W_proj   [4096 x 6144]
  gemm_f16_wmma<true><<<dim3(3 * D_MODEL / 128, BS / 256), 256, 0, stream>>>(
      xh, Wph, qkvh, (int)BS, 3 * D_MODEL, D_MODEL);
  // 3) RoPE + head split
  rope_split<<<(B_SZ * N_HEAD * S_LEN * 64) / 256, 256, 0, stream>>>(qkvh, Qr, Kr, Vr);
  // 4) causal flash attention
  flash_attn<<<dim3(S_LEN / 64, B_SZ * N_HEAD), 128, 0, stream>>>(Qr, Kr, Vr, Obuf);
  // 5) out = O @ W_out    [4096 x 2048]
  gemm_f16_wmma<false><<<dim3(D_MODEL / 128, BS / 256), 256, 0, stream>>>(
      Obuf, Woh, out, (int)BS, D_MODEL, D_MODEL);
}
